// GraphAttentionLayer_68126771249441
// MI455X (gfx1250) — matrix operand order repaired and checked by reading, not compiled, not run on hardware
//
#include <hip/hip_runtime.h>

#define N_NODES 8192
#define F       64
#define ALPHA   0.2f
#define NEG_INF (-9.0e15f)
#define KSPLIT  4
#define KCHUNK  (N_NODES / KSPLIT)   // 2048

typedef __attribute__((ext_vector_type(16))) __bf16 v16bf;
typedef __attribute__((ext_vector_type(8)))  float  v8f;

__device__ __forceinline__ float4 ld4(const float* p) {
    return *reinterpret_cast<const float4*>(p);
}
__device__ __forceinline__ __bf16 sgnbf(float v) {
    return (__bf16)((v > 0.f) ? 1.f : ((v < 0.f) ? -1.f : 0.f));
}
// B-fragment: lane (half,n) reads K = k0+8*half+[0,8) and k0+16+8*half+[0,8) = two aligned 16B loads.
__device__ __forceinline__ v16bf bload(const __bf16* baseT, int nb, int n, int k0, int half) {
    typedef __attribute__((ext_vector_type(8))) __bf16 v8bf_;
    const __bf16* p_ = baseT + (size_t)(nb * 16 + n) * N_NODES + k0 + 8 * half;
    v8bf_ lo_ = *reinterpret_cast<const v8bf_*>(p_);
    v8bf_ hi_ = *reinterpret_cast<const v8bf_*>(p_ + 16);
    return __builtin_shufflevector(lo_, hi_, 0, 1, 2, 3, 4, 5, 6, 7, 8, 9, 10, 11, 12, 13, 14, 15);
}

// ---------------------------------------------------------------------------
// K1: h_att = h @ W_att (fp32 exact, tiny). Emit bf16 transposed copies
//     h_attT[64][8192] and hT[64][8192] for the WMMA B operands.
// ---------------------------------------------------------------------------
__global__ void k1_hatt(const float* __restrict__ h, const float* __restrict__ W,
                        __bf16* __restrict__ hattT, __bf16* __restrict__ hT) {
    __shared__ float sW[F * F];
    __shared__ float sh[16 * F];
    const int t  = threadIdx.x;
    const int r0 = blockIdx.x * 16;
#pragma unroll
    for (int i = 0; i < 16; ++i) sW[t + i * 256] = W[t + i * 256];
#pragma unroll
    for (int i = 0; i < 4; ++i) {
        int lin = t + i * 256;
        int r = lin >> 6, f = lin & 63;
        float v = h[(size_t)(r0 + r) * F + f];
        sh[lin] = v;
        hT[(size_t)f * N_NODES + r0 + r] = (__bf16)v;
    }
    __syncthreads();
#pragma unroll
    for (int i = 0; i < 4; ++i) {
        int lin = t + i * 256;
        int r = lin >> 6, j = lin & 63;
        float acc = 0.f;
#pragma unroll
        for (int f = 0; f < F; ++f) acc += sh[r * F + f] * sW[f * F + j];
        hattT[(size_t)j * N_NODES + r0 + r] = (__bf16)acc;
    }
}

// ---------------------------------------------------------------------------
// K2: partial s = sign(node_adj[:, kbeg:kend]) @ h_att, split-K x4,
//     one wave32 per (16-row slab, K-chunk), software-pipelined WMMA loop.
// ---------------------------------------------------------------------------
__global__ void __launch_bounds__(32)
k2_partial(const float* __restrict__ node_adj, const __bf16* __restrict__ hattT,
           float* __restrict__ P) {
    const int lane = threadIdx.x;
    const int half = lane >> 4;
    const int n    = lane & 15;
    const int slab = blockIdx.x >> 2;
    const int kc   = blockIdx.x & 3;
    const int kbeg = kc * KCHUNK, kend = kbeg + KCHUNK;
    const float* rowp = node_adj + (size_t)(slab * 16 + n) * N_NODES;

    v8f acc[4];
#pragma unroll
    for (int nb = 0; nb < 4; ++nb) acc[nb] = (v8f){0, 0, 0, 0, 0, 0, 0, 0};

    // pipeline prologue: tile 0 in flight
    int cb1 = kbeg + 8 * half, cb2 = cb1 + 16;
    float4 x0 = ld4(rowp + cb1), x1 = ld4(rowp + cb1 + 4);
    float4 y0 = ld4(rowp + cb2), y1 = ld4(rowp + cb2 + 4);
    v16bf B0 = bload(hattT, 0, n, kbeg, half);
    v16bf B1 = bload(hattT, 1, n, kbeg, half);
    v16bf B2 = bload(hattT, 2, n, kbeg, half);
    v16bf B3 = bload(hattT, 3, n, kbeg, half);

    for (int k0 = kbeg; k0 < kend; k0 += 32) {
        // issue next tile's loads (dummy re-load of kbeg on the last trip)
        const int kn  = (k0 + 32 < kend) ? (k0 + 32) : kbeg;
        const int nb1 = kn + 8 * half, nb2 = nb1 + 16;
        float4 X0 = ld4(rowp + nb1), X1 = ld4(rowp + nb1 + 4);
        float4 Y0 = ld4(rowp + nb2), Y1 = ld4(rowp + nb2 + 4);
        v16bf N0 = bload(hattT, 0, n, kn, half);
        v16bf N1 = bload(hattT, 1, n, kn, half);
        v16bf N2 = bload(hattT, 2, n, kn, half);
        v16bf N3 = bload(hattT, 3, n, kn, half);
        int kp = k0 + 512; kp = (kp < kend - 32) ? kp : k0;   // branchless
        __builtin_prefetch(rowp + kp, 0, 1);
        // consume current tile
        v16bf A;
        A[0]  = sgnbf(x0.x); A[1]  = sgnbf(x0.y); A[2]  = sgnbf(x0.z); A[3]  = sgnbf(x0.w);
        A[4]  = sgnbf(x1.x); A[5]  = sgnbf(x1.y); A[6]  = sgnbf(x1.z); A[7]  = sgnbf(x1.w);
        A[8]  = sgnbf(y0.x); A[9]  = sgnbf(y0.y); A[10] = sgnbf(y0.z); A[11] = sgnbf(y0.w);
        A[12] = sgnbf(y1.x); A[13] = sgnbf(y1.y); A[14] = sgnbf(y1.z); A[15] = sgnbf(y1.w);
        acc[0] = __builtin_amdgcn_wmma_f32_16x16x32_bf16(false, A, false, B0, (short)0, acc[0], false, false);
        acc[1] = __builtin_amdgcn_wmma_f32_16x16x32_bf16(false, A, false, B1, (short)0, acc[1], false, false);
        acc[2] = __builtin_amdgcn_wmma_f32_16x16x32_bf16(false, A, false, B2, (short)0, acc[2], false, false);
        acc[3] = __builtin_amdgcn_wmma_f32_16x16x32_bf16(false, A, false, B3, (short)0, acc[3], false, false);
        x0 = X0; x1 = X1; y0 = Y0; y1 = Y1;
        B0 = N0; B1 = N1; B2 = N2; B3 = N3;
    }

    const int row0 = slab * 16;
#pragma unroll
    for (int nb = 0; nb < 4; ++nb)
#pragma unroll
        for (int v = 0; v < 8; ++v)
            P[((size_t)kc * N_NODES + row0 + v + 8 * half) * F + nb * 16 + n] = acc[nb][v];
}

// ---------------------------------------------------------------------------
// K2b: e[r] = leaky_relu( sum_f |sum_kc P| * a[f] ). One wave per row.
// ---------------------------------------------------------------------------
__global__ void __launch_bounds__(32)
k2b_escore(const float* __restrict__ P, const float* __restrict__ a_vec,
           float* __restrict__ e_buf) {
    const int r = blockIdx.x, t = threadIdx.x;
    float sum = 0.f;
#pragma unroll
    for (int i = 0; i < 2; ++i) {
        int f = t + i * 32;
        float s = 0.f;
#pragma unroll
        for (int kc = 0; kc < KSPLIT; ++kc) s += P[((size_t)kc * N_NODES + r) * F + f];
        sum += fabsf(s) * a_vec[f];
    }
#pragma unroll
    for (int off = 16; off > 0; off >>= 1) sum += __shfl_xor(sum, off, 32);
    if (t == 0) e_buf[r] = (sum > 0.f) ? sum : ALPHA * sum;
}

// ---------------------------------------------------------------------------
// K3: per-row masked-softmax stats over edge_adj (streaming pass 1).
// ---------------------------------------------------------------------------
__global__ void k3_maxsum(const float* __restrict__ edge_adj, const float* __restrict__ e_buf,
                          float* __restrict__ m_buf, float* __restrict__ s_buf) {
    __shared__ float red[256];
    const int row = blockIdx.x, t = threadIdx.x;
    const float* rp = edge_adj + (size_t)row * N_NODES;
    float sc[32];
    float mx = NEG_INF;
#pragma unroll
    for (int i = 0; i < 32; ++i) {
        int j = t + i * 256;
        float v = rp[j];
        float s = (v != 0.f) ? e_buf[j] : NEG_INF;
        sc[i] = s;
        mx = fmaxf(mx, s);
    }
    red[t] = mx;
    __syncthreads();
    for (int o = 128; o > 0; o >>= 1) {
        if (t < o) red[t] = fmaxf(red[t], red[t + o]);
        __syncthreads();
    }
    float M = red[0];
    __syncthreads();
    float sm = 0.f;
#pragma unroll
    for (int i = 0; i < 32; ++i) sm += __expf(sc[i] - M);
    red[t] = sm;
    __syncthreads();
    for (int o = 128; o > 0; o >>= 1) {
        if (t < o) red[t] += red[t + o];
        __syncthreads();
    }
    if (t == 0) { m_buf[row] = M; s_buf[row] = red[0]; }
}

// ---------------------------------------------------------------------------
// K4: fused normalize + attention store + partial h_prime (split-K x4),
//     software-pipelined bf16 WMMA.
// ---------------------------------------------------------------------------
__global__ void __launch_bounds__(32)
k4_attn(const float* __restrict__ edge_adj, const __bf16* __restrict__ hT,
        const float* __restrict__ e_buf, const float* __restrict__ m_buf,
        const float* __restrict__ s_buf, float* __restrict__ Q,
        float* __restrict__ att_out) {
    const int lane = threadIdx.x;
    const int half = lane >> 4;
    const int n    = lane & 15;
    const int slab = blockIdx.x >> 2;
    const int kc   = blockIdx.x & 3;
    const int kbeg = kc * KCHUNK, kend = kbeg + KCHUNK;
    const int row  = slab * 16 + n;
    const float* rp = edge_adj + (size_t)row * N_NODES;
    float*       ap = att_out  + (size_t)row * N_NODES;
    const float mrow  = m_buf[row];
    const float inv_s = 1.f / s_buf[row];

    v8f acc[4];
#pragma unroll
    for (int nb = 0; nb < 4; ++nb) acc[nb] = (v8f){0, 0, 0, 0, 0, 0, 0, 0};

    int cb1 = kbeg + 8 * half, cb2 = cb1 + 16;
    float4 x0 = ld4(rp + cb1), x1 = ld4(rp + cb1 + 4);
    float4 y0 = ld4(rp + cb2), y1 = ld4(rp + cb2 + 4);
    float4 e0 = ld4(e_buf + cb1), e1 = ld4(e_buf + cb1 + 4);
    float4 e2 = ld4(e_buf + cb2), e3 = ld4(e_buf + cb2 + 4);
    v16bf B0 = bload(hT, 0, n, kbeg, half);
    v16bf B1 = bload(hT, 1, n, kbeg, half);
    v16bf B2 = bload(hT, 2, n, kbeg, half);
    v16bf B3 = bload(hT, 3, n, kbeg, half);

    for (int k0 = kbeg; k0 < kend; k0 += 32) {
        const int kn  = (k0 + 32 < kend) ? (k0 + 32) : kbeg;
        const int nb1 = kn + 8 * half, nb2 = nb1 + 16;
        float4 X0 = ld4(rp + nb1), X1 = ld4(rp + nb1 + 4);
        float4 Y0 = ld4(rp + nb2), Y1 = ld4(rp + nb2 + 4);
        float4 E0 = ld4(e_buf + nb1), E1 = ld4(e_buf + nb1 + 4);
        float4 E2 = ld4(e_buf + nb2), E3 = ld4(e_buf + nb2 + 4);
        v16bf N0 = bload(hT, 0, n, kn, half);
        v16bf N1 = bload(hT, 1, n, kn, half);
        v16bf N2 = bload(hT, 2, n, kn, half);
        v16bf N3 = bload(hT, 3, n, kn, half);
        int kp = k0 + 512; kp = (kp < kend - 32) ? kp : k0;
        __builtin_prefetch(rp + kp, 0, 1);

        const int b1 = k0 + 8 * half, b2 = b1 + 16;
        float av[16];
        v16bf A;
#define AW(ev, ej, i)                                                      \
        { float sc_ = ((ev) != 0.f) ? (ej) : NEG_INF;                      \
          float at_ = __expf(sc_ - mrow) * inv_s;                          \
          av[i] = at_; A[i] = (__bf16)(at_ * (ev)); }
        AW(x0.x, e0.x, 0)  AW(x0.y, e0.y, 1)  AW(x0.z, e0.z, 2)  AW(x0.w, e0.w, 3)
        AW(x1.x, e1.x, 4)  AW(x1.y, e1.y, 5)  AW(x1.z, e1.z, 6)  AW(x1.w, e1.w, 7)
        AW(y0.x, e2.x, 8)  AW(y0.y, e2.y, 9)  AW(y0.z, e2.z, 10) AW(y0.w, e2.w, 11)
        AW(y1.x, e3.x, 12) AW(y1.y, e3.y, 13) AW(y1.z, e3.z, 14) AW(y1.w, e3.w, 15)
#undef AW
        *reinterpret_cast<float4*>(ap + b1)     = make_float4(av[0],  av[1],  av[2],  av[3]);
        *reinterpret_cast<float4*>(ap + b1 + 4) = make_float4(av[4],  av[5],  av[6],  av[7]);
        *reinterpret_cast<float4*>(ap + b2)     = make_float4(av[8],  av[9],  av[10], av[11]);
        *reinterpret_cast<float4*>(ap + b2 + 4) = make_float4(av[12], av[13], av[14], av[15]);
        acc[0] = __builtin_amdgcn_wmma_f32_16x16x32_bf16(false, A, false, B0, (short)0, acc[0], false, false);
        acc[1] = __builtin_amdgcn_wmma_f32_16x16x32_bf16(false, A, false, B1, (short)0, acc[1], false, false);
        acc[2] = __builtin_amdgcn_wmma_f32_16x16x32_bf16(false, A, false, B2, (short)0, acc[2], false, false);
        acc[3] = __builtin_amdgcn_wmma_f32_16x16x32_bf16(false, A, false, B3, (short)0, acc[3], false, false);
        x0 = X0; x1 = X1; y0 = Y0; y1 = Y1;
        e0 = E0; e1 = E1; e2 = E2; e3 = E3;
        B0 = N0; B1 = N1; B2 = N2; B3 = N3;
    }

    const int row0 = slab * 16;
#pragma unroll
    for (int nb = 0; nb < 4; ++nb)
#pragma unroll
        for (int v = 0; v < 8; ++v)
            Q[((size_t)kc * N_NODES + row0 + v + 8 * half) * F + nb * 16 + n] = acc[nb][v];
}

// ---------------------------------------------------------------------------
// K4b: h_prime = sum over K-chunks of partials (fixed order, deterministic).
// ---------------------------------------------------------------------------
__global__ void k4b_reduce(const float* __restrict__ Q, float* __restrict__ h_prime) {
    size_t i = (size_t)blockIdx.x * 256 + threadIdx.x;
    float s = 0.f;
#pragma unroll
    for (int kc = 0; kc < KSPLIT; ++kc) s += Q[(size_t)kc * N_NODES * F + i];
    h_prime[i] = s;
}

// ---------------------------------------------------------------------------
extern "C" void kernel_launch(void* const* d_in, const int* in_sizes, int n_in,
                              void* d_out, int out_size, void* d_ws, size_t ws_size,
                              hipStream_t stream) {
    const float* h        = (const float*)d_in[0];
    const float* node_adj = (const float*)d_in[1];
    const float* edge_adj = (const float*)d_in[2];
    const float* W_att    = (const float*)d_in[3];
    const float* a_vec    = (const float*)d_in[4];

    float* h_prime = (float*)d_out;                         // [8192, 64]
    float* att_out = h_prime + (size_t)N_NODES * F;         // [8192, 8192]

    char* ws = (char*)d_ws;
    __bf16* hattT = (__bf16*)ws;                            // 1 MB
    __bf16* hT    = (__bf16*)(ws + (1u << 20));             // 1 MB
    float*  e_buf = (float*)(ws + (2u << 20));              // [8192]
    float*  m_buf = e_buf + N_NODES;
    float*  s_buf = m_buf + N_NODES;
    float*  Pbuf  = (float*)(ws + (4u << 20));              // 8 MB, reused K2->K4

    k1_hatt   <<<N_NODES / 16,          256, 0, stream>>>(h, W_att, hattT, hT);
    k2_partial<<<(N_NODES / 16) * KSPLIT, 32, 0, stream>>>(node_adj, hattT, Pbuf);
    k2b_escore<<<N_NODES,                32, 0, stream>>>(Pbuf, a_vec, e_buf);
    k3_maxsum <<<N_NODES,               256, 0, stream>>>(edge_adj, e_buf, m_buf, s_buf);
    k4_attn   <<<(N_NODES / 16) * KSPLIT, 32, 0, stream>>>(edge_adj, hT, e_buf, m_buf,
                                                           s_buf, Pbuf, att_out);
    k4b_reduce<<<(N_NODES * F) / 256,   256, 0, stream>>>(Pbuf, h_prime);
}
